// RCNNNet_51960514347529
// MI455X (gfx1250) — compile-verified, matrix operand order checked by reading, not run on hardware
//
#include <hip/hip_runtime.h>
#include <hip/hip_bf16.h>

typedef __attribute__((ext_vector_type(16))) _Float16 v16h;
typedef __attribute__((ext_vector_type(8)))  _Float16 v8h;
typedef __attribute__((ext_vector_type(8)))  float    v8f;

// ---------------------------------------------------------------------------
// Wave-level GEMM, compile-time shaped: each wave (32 lanes) owns 16 rows of
// the activation tile in LDS and produces NT*16 output columns via
// v_wmma_f32_16x16x32_f16.  All KC A-operands are hoisted into registers
// BEFORE any LDS store, so the compiler needs no LDS alias analysis and emits
// each ds_load_b128 exactly once per layer.
//   bufA: LDS activations, strideA halves (multiple of 8 -> 16B aligned loads)
//   Wpk : global weights pre-packed to WMMA B-operand layout
//         index = ((kc*NT + nt)*32 + lane)*16 + j
// A-operand layout (16-bit A 16x32): lane l covers row M=l&15; element j maps
// to K = (j&7) + 16*(j>>3) + 8*(l>>4)  -> two contiguous 8-half segments.
// D layout: lane l covers N=l&15, VGPR v covers M = v + 8*(l>>4).
// ---------------------------------------------------------------------------
template <int KC, int NT, bool RELU>
__device__ __forceinline__ void wave_gemm_t(
    const _Float16* bufA, int strideA,
    _Float16* bufO, int strideO,
    const _Float16* __restrict__ Wpk,
    const float* __restrict__ bias, int biasN)
{
    const int lane = threadIdx.x & 31;
    const int wave = threadIdx.x >> 5;
    const int rowBase = wave * 16;
    const int hi = lane >> 4;
    const int m  = lane & 15;
    const _Float16* aRow = bufA + (size_t)(rowBase + m) * strideA;

    v16h aReg[KC];
#pragma unroll
    for (int kc = 0; kc < KC; ++kc) {
        v8h a0 = *(const v8h*)(aRow + kc * 32 + hi * 8);
        v8h a1 = *(const v8h*)(aRow + kc * 32 + 16 + hi * 8);
        aReg[kc] = __builtin_shufflevector(a0, a1,
                      0,1,2,3,4,5,6,7,8,9,10,11,12,13,14,15);
    }

    for (int nt = 0; nt < NT; ++nt) {
        v8f acc = {};
        const _Float16* wp = Wpk + ((size_t)nt * 32 + lane) * 16;
#pragma unroll
        for (int kc = 0; kc < KC; ++kc) {
            v16h b = *(const v16h*)(wp + (size_t)kc * NT * 512);
            acc = __builtin_amdgcn_wmma_f32_16x16x32_f16(
                      false, aReg[kc], false, b, (short)0, acc, false, false);
        }
        int col = nt * 16 + m;
        float bv = (col < biasN) ? bias[col] : 0.0f;
#pragma unroll
        for (int v = 0; v < 8; ++v) {
            float x = acc[v] + bv;
            if (RELU) x = fmaxf(x, 0.0f);
            bufO[(size_t)(rowBase + v + 8 * hi) * strideO + col] = (_Float16)x;
        }
    }
}

// ---------------------------------------------------------------------------
// Pack (K x N) f32 weights into WMMA B-operand f16 layout, zero-padded.
// ---------------------------------------------------------------------------
__global__ void pack_w_kernel(const float* __restrict__ W, int K, int N,
                              int kChunks, int nTiles, _Float16* __restrict__ dst)
{
    int idx = blockIdx.x * blockDim.x + threadIdx.x;
    int total = kChunks * nTiles * 512;
    if (idx >= total) return;
    int j    = idx & 15;
    int lane = (idx >> 4) & 31;
    int t    = (idx >> 9) % nTiles;
    int kc   = (idx >> 9) / nTiles;
    int n = t * 16 + (lane & 15);
    int k = kc * 32 + (j & 7) + ((j >> 3) << 4) + ((lane >> 4) << 3);
    float v = (k < K && n < N) ? W[(size_t)k * N + n] : 0.0f;
    dst[idx] = (_Float16)v;
}

// ---------------------------------------------------------------------------
// Per-batch: mean-center xyz, depth feature -> feat4 (B,N,4) f16.
// ---------------------------------------------------------------------------
__global__ void prep_kernel(const float* __restrict__ xyz, _Float16* __restrict__ feat4)
{
    __shared__ float red[3 * 256];
    const int b = blockIdx.x, tid = threadIdx.x;
    float sx = 0.f, sy = 0.f, sz = 0.f;
    for (int p = tid; p < 512; p += 256) {
        const float* q = xyz + ((size_t)b * 512 + p) * 3;
        sx += q[0]; sy += q[1]; sz += q[2];
    }
    red[tid] = sx; red[256 + tid] = sy; red[512 + tid] = sz;
    __syncthreads();
    for (int off = 128; off > 0; off >>= 1) {
        if (tid < off) {
            red[tid]       += red[tid + off];
            red[256 + tid] += red[256 + tid + off];
            red[512 + tid] += red[512 + tid + off];
        }
        __syncthreads();
    }
    float mx = red[0] / 512.f, my = red[256] / 512.f, mz = red[512] / 512.f;
    for (int p = tid; p < 512; p += 256) {
        const float* q = xyz + ((size_t)b * 512 + p) * 3;
        float x = q[0], y = q[1], z = q[2];
        float d = sqrtf(x * x + y * y + z * z) / 70.f - 0.5f;
        _Float16* o = feat4 + ((size_t)b * 512 + p) * 4;
        o[0] = (_Float16)(x - mx);
        o[1] = (_Float16)(y - my);
        o[2] = (_Float16)(z - mz);
        o[3] = (_Float16)d;
    }
}

// ---------------------------------------------------------------------------
// Fused xyz_up (4->128->128) + merge (256->128). 128 rows / WG, 8 waves.
// LDS: bufA 128x264 (staging + wide input), bufB 128x136.
// ---------------------------------------------------------------------------
__global__ void up_merge_kernel(
    const _Float16* __restrict__ feat4, const float* __restrict__ features,
    const _Float16* Wu1, const float* bu1,
    const _Float16* Wu2, const float* bu2,
    const _Float16* Wm,  const float* bm_,
    _Float16* __restrict__ merged)
{
    extern __shared__ char smemRaw[];
    _Float16* bufA = (_Float16*)smemRaw;                       // 128 x 264
    _Float16* bufB = (_Float16*)(smemRaw + 128 * 264 * 2);     // 128 x 136
    const int tid = threadIdx.x;
    const size_t rowG = (size_t)blockIdx.x * 128;

    for (int e = tid; e < 128 * 32; e += 256) {                // stage feat4, pad K->32
        int r = e >> 5, k = e & 31;
        bufA[r * 264 + k] = (k < 4) ? feat4[(rowG + r) * 4 + k] : (_Float16)0.0f;
    }
    __syncthreads();
    wave_gemm_t<1, 8, true>(bufA, 264, bufB, 136, Wu1, bu1, 128);
    __syncthreads();
    wave_gemm_t<4, 8, true>(bufB, 136, bufA, 264, Wu2, bu2, 128);
    __syncthreads();
    for (int e = tid; e < 128 * 128; e += 256) {               // concat raw features
        int r = e >> 7, k = e & 127;
        bufA[r * 264 + 128 + k] = (_Float16)features[(rowG + r) * 128 + k];
    }
    __syncthreads();
    wave_gemm_t<8, 8, true>(bufA, 264, bufB, 136, Wm, bm_, 128);
    __syncthreads();
    for (int e = tid; e < 128 * 128; e += 256) {
        int r = e >> 7, k = e & 127;
        merged[(rowG + r) * 128 + k] = bufB[r * 136 + k];
    }
}

// ---------------------------------------------------------------------------
// Farthest point sampling, one wave per batch, dist/points cached in LDS.
// ---------------------------------------------------------------------------
__global__ void fps_kernel(const float* __restrict__ xyz, int Npts, int npoint,
                           float* __restrict__ newxyz)
{
    __shared__ float px[512 * 3];
    __shared__ float dist[512];
    const int b = blockIdx.x, lane = threadIdx.x;
    for (int p = lane; p < Npts; p += 32) {
        px[p * 3 + 0] = xyz[((size_t)b * Npts + p) * 3 + 0];
        px[p * 3 + 1] = xyz[((size_t)b * Npts + p) * 3 + 1];
        px[p * 3 + 2] = xyz[((size_t)b * Npts + p) * 3 + 2];
        dist[p] = 1e10f;
    }
    int far = 0;
    for (int it = 0; it < npoint; ++it) {
        if (lane == 0) {
            newxyz[((size_t)b * npoint + it) * 3 + 0] = px[far * 3 + 0];
            newxyz[((size_t)b * npoint + it) * 3 + 1] = px[far * 3 + 1];
            newxyz[((size_t)b * npoint + it) * 3 + 2] = px[far * 3 + 2];
        }
        float fx = px[far * 3 + 0], fy = px[far * 3 + 1], fz = px[far * 3 + 2];
        float best = -1.0f; int bi = 0;
        for (int p = lane; p < Npts; p += 32) {
            float dx = px[p * 3 + 0] - fx, dy = px[p * 3 + 1] - fy, dz = px[p * 3 + 2] - fz;
            float nd = fminf(dist[p], dx * dx + dy * dy + dz * dz);
            dist[p] = nd;
            if (nd > best) { best = nd; bi = p; }
        }
        for (int off = 16; off > 0; off >>= 1) {
            float ob = __shfl_xor(best, off, 32);
            int   oi = __shfl_xor(bi, off, 32);
            if (ob > best || (ob == best && oi < bi)) { best = ob; bi = oi; }
        }
        far = bi;
    }
}

// ---------------------------------------------------------------------------
// Ball query: one thread per (batch, center); first nsample in-radius indices
// in ascending order, padded with first hit (or Npts-1 if none).
// ---------------------------------------------------------------------------
__global__ void ballq_kernel(const float* __restrict__ xyz, int Npts,
                             const float* __restrict__ newxyz, int ncenter,
                             float r2, int* __restrict__ gidx)
{
    int t = blockIdx.x * blockDim.x + threadIdx.x;
    int total = 128 * ncenter;
    if (t >= total) return;
    int b = t / ncenter, c = t - b * ncenter;
    float cx = newxyz[((size_t)b * ncenter + c) * 3 + 0];
    float cy = newxyz[((size_t)b * ncenter + c) * 3 + 1];
    float cz = newxyz[((size_t)b * ncenter + c) * 3 + 2];
    int* out = gidx + ((size_t)b * ncenter + c) * 64;
    int cnt = 0;
    for (int p = 0; p < Npts && cnt < 64; ++p) {
        float dx = xyz[((size_t)b * Npts + p) * 3 + 0] - cx;
        float dy = xyz[((size_t)b * Npts + p) * 3 + 1] - cy;
        float dz = xyz[((size_t)b * Npts + p) * 3 + 2] - cz;
        if (dx * dx + dy * dy + dz * dz < r2) out[cnt++] = p;
    }
    int fill = (cnt > 0) ? out[0] : (Npts - 1);
    for (; cnt < 64; ++cnt) out[cnt] = fill;
}

// ---------------------------------------------------------------------------
// Fused SA stage: gather (3 rel-xyz + 128 feat -> K=131 pad 160), 3 WMMA
// layers fully in LDS, max-pool over 64 samples. 2 centers (128 rows) per WG.
// N3T = layer-3 column tiles (8 -> 128ch for SA1, 16 -> 256ch for SA2).
// ---------------------------------------------------------------------------
template <int N3T>
__global__ void sa_mlp_kernel(
    const float* __restrict__ xyz, int Npts,
    const _Float16* __restrict__ feat,
    const float* __restrict__ newxyz, const int* __restrict__ gidx, int ncenter,
    const _Float16* W1, const float* b1,
    const _Float16* W2, const float* b2,
    const _Float16* W3, const float* b3,
    _Float16* __restrict__ outFeat)
{
    extern __shared__ char smemRaw[];
    _Float16* bufA = (_Float16*)smemRaw;                          // 128 x 168
    _Float16* bufB = (_Float16*)(smemRaw + 128 * 168 * 2);        // 128 x 264
    int*   sIdx = (int*)(smemRaw + 128 * 168 * 2 + 128 * 264 * 2);
    float* sCen = (float*)(sIdx + 128);
    const int tid = threadIdx.x;
    const int pairPerB = ncenter >> 1;
    const int b  = blockIdx.x / pairPerB;
    const int c0 = (blockIdx.x % pairPerB) * 2;

    if (tid < 128) {
        int ci = c0 + (tid >> 6), s = tid & 63;
        sIdx[tid] = gidx[((size_t)b * ncenter + ci) * 64 + s];
    }
    if (tid < 6) sCen[tid] = newxyz[((size_t)b * ncenter + c0) * 3 + tid];
    __syncthreads();

    for (int e = tid; e < 128 * 160; e += 256) {
        int r = e / 160, k = e - r * 160;
        int ci = r >> 6;
        int p = sIdx[r];
        _Float16 v;
        if (k < 3)        v = (_Float16)(xyz[((size_t)b * Npts + p) * 3 + k] - sCen[ci * 3 + k]);
        else if (k < 131) v = feat[((size_t)b * Npts + p) * 128 + (k - 3)];
        else              v = (_Float16)0.0f;
        bufA[r * 168 + k] = v;
    }
    __syncthreads();
    wave_gemm_t<5, 8, true>(bufA, 168, bufB, 264, W1, b1, 128);
    __syncthreads();
    wave_gemm_t<4, 8, true>(bufB, 264, bufA, 168, W2, b2, 128);
    __syncthreads();
    wave_gemm_t<4, N3T, true>(bufA, 168, bufB, 264, W3, b3, N3T * 16);
    __syncthreads();

    const int outW = N3T * 16;
    for (int e = tid; e < 2 * outW; e += 256) {
        int ci = e / outW, k = e - ci * outW;
        float mx = -1e30f;
        for (int s = 0; s < 64; ++s)
            mx = fmaxf(mx, (float)bufB[(ci * 64 + s) * 264 + k]);
        outFeat[(((size_t)b * ncenter) + c0 + ci) * outW + k] = (_Float16)mx;
    }
}

// ---------------------------------------------------------------------------
// SA3: concat(xyz2, feat2) K=259 pad 288 -> 256 -> 256 -> 512, max over 32.
// 4 batches (128 rows) per WG. LDS: bufA 128x296, bufB 128x520.
// ---------------------------------------------------------------------------
__global__ void sa3_kernel(
    const float* __restrict__ newxyz2, const _Float16* __restrict__ feat2,
    const _Float16* W1, const float* b1,
    const _Float16* W2, const float* b2,
    const _Float16* W3, const float* b3,
    _Float16* __restrict__ g)
{
    extern __shared__ char smemRaw[];
    _Float16* bufA = (_Float16*)smemRaw;                        // 128 x 296
    _Float16* bufB = (_Float16*)(smemRaw + 128 * 296 * 2);      // 128 x 520
    const int tid = threadIdx.x;
    const int b0 = blockIdx.x * 4;

    for (int e = tid; e < 128 * 296; e += 256) {
        int r = e / 296, k = e - r * 296;
        int bb = b0 + (r >> 5), c = r & 31;
        _Float16 v;
        if (k < 3)        v = (_Float16)newxyz2[((size_t)bb * 32 + c) * 3 + k];
        else if (k < 259) v = feat2[((size_t)bb * 32 + c) * 256 + (k - 3)];
        else              v = (_Float16)0.0f;
        bufA[r * 296 + k] = v;
    }
    __syncthreads();
    wave_gemm_t<9, 16, true>(bufA, 296, bufB, 520, W1, b1, 256);
    __syncthreads();
    wave_gemm_t<8, 16, true>(bufB, 520, bufA, 296, W2, b2, 256);
    __syncthreads();
    wave_gemm_t<8, 32, true>(bufA, 296, bufB, 520, W3, b3, 512);
    __syncthreads();
    for (int e = tid; e < 4 * 512; e += 256) {
        int q = e >> 9, k = e & 511;
        float mx = -1e30f;
        for (int c = 0; c < 32; ++c)
            mx = fmaxf(mx, (float)bufB[(q * 32 + c) * 520 + k]);
        g[(size_t)(b0 + q) * 512 + k] = (_Float16)mx;
    }
}

// ---------------------------------------------------------------------------
// Heads: both 512->256->256->{1,70} MLPs on all 128 rows in one WG.
// LDS: bufA 128x520 (keeps g), bufB/bufC 128x264.
// ---------------------------------------------------------------------------
__global__ void heads_kernel(
    const _Float16* __restrict__ g,
    const _Float16* Wc1, const float* bc1,
    const _Float16* Wc2, const float* bc2,
    const _Float16* Wc3, const float* bc3,
    const _Float16* Wr1, const float* br1,
    const _Float16* Wr2, const float* br2,
    const _Float16* Wr3, const float* br3,
    float* __restrict__ out)
{
    extern __shared__ char smemRaw[];
    _Float16* bufA = (_Float16*)smemRaw;                                     // 128 x 520
    _Float16* bufB = (_Float16*)(smemRaw + 128 * 520 * 2);                   // 128 x 264
    _Float16* bufC = (_Float16*)(smemRaw + 128 * 520 * 2 + 128 * 264 * 2);   // 128 x 264
    const int tid = threadIdx.x;

    for (int e = tid; e < 128 * 512; e += 256) {
        int r = e >> 9, k = e & 511;
        bufA[r * 520 + k] = g[e];
    }
    __syncthreads();
    // cls head
    wave_gemm_t<16, 16, true>(bufA, 520, bufB, 264, Wc1, bc1, 256);
    __syncthreads();
    wave_gemm_t<8, 16, true>(bufB, 264, bufC, 264, Wc2, bc2, 256);
    __syncthreads();
    wave_gemm_t<8, 1, false>(bufC, 264, bufB, 264, Wc3, bc3, 1);
    __syncthreads();
    if (tid < 128) out[tid] = (float)bufB[tid * 264 + 0];
    __syncthreads();
    // reg head
    wave_gemm_t<16, 16, true>(bufA, 520, bufB, 264, Wr1, br1, 256);
    __syncthreads();
    wave_gemm_t<8, 16, true>(bufB, 264, bufC, 264, Wr2, br2, 256);
    __syncthreads();
    wave_gemm_t<8, 5, false>(bufC, 264, bufB, 264, Wr3, br3, 70);
    __syncthreads();
    for (int e = tid; e < 128 * 70; e += 256) {
        int r = e / 70, k = e - r * 70;
        out[128 + e] = (float)bufB[r * 264 + k];
    }
}

// ---------------------------------------------------------------------------
// Host side.
// Input order (dict keys sorted by jax tree flatten):
//  0 xyz, 1 features, then params: cls(W1,b1,W2,b2,W3,b3)=2..7, merge=8..9,
//  reg=10..15, sa1=16..21, sa2=22..27, sa3=28..33, xyz_up=34..37.
// ---------------------------------------------------------------------------
extern "C" void kernel_launch(void* const* d_in, const int* in_sizes, int n_in,
                              void* d_out, int out_size, void* d_ws, size_t ws_size,
                              hipStream_t stream)
{
    (void)in_sizes; (void)n_in; (void)out_size; (void)ws_size;
    const int B = 128, N = 512;
    char* ws = (char*)d_ws;
    size_t off = 0;
    auto alloc = [&](size_t bytes) -> void* {
        off = (off + 255) & ~(size_t)255;
        void* p = ws + off;
        off += bytes;
        return p;
    };

    _Float16* feat4  = (_Float16*)alloc((size_t)B * N * 4 * 2);
    _Float16* merged = (_Float16*)alloc((size_t)B * N * 128 * 2);
    float*    nxyz1  = (float*)alloc((size_t)B * 128 * 3 * 4);
    int*      gidx1  = (int*)alloc((size_t)B * 128 * 64 * 4);
    _Float16* feat1  = (_Float16*)alloc((size_t)B * 128 * 128 * 2);
    float*    nxyz2  = (float*)alloc((size_t)B * 32 * 3 * 4);
    int*      gidx2  = (int*)alloc((size_t)B * 32 * 64 * 4);
    _Float16* feat2  = (_Float16*)alloc((size_t)B * 32 * 256 * 2);
    _Float16* gbuf   = (_Float16*)alloc((size_t)B * 512 * 2);

    auto packW = [&](int srcIdx, int K, int Ncol) -> _Float16* {
        int kch = (K + 31) / 32, ntl = (Ncol + 15) / 16;
        int total = kch * ntl * 512;
        _Float16* dst = (_Float16*)alloc((size_t)total * 2);
        pack_w_kernel<<<(total + 255) / 256, 256, 0, stream>>>(
            (const float*)d_in[srcIdx], K, Ncol, kch, ntl, dst);
        return dst;
    };
    auto bias = [&](int i) { return (const float*)d_in[i]; };

    _Float16* pUp1 = packW(34,   4, 128);
    _Float16* pUp2 = packW(36, 128, 128);
    _Float16* pMg  = packW( 8, 256, 128);
    _Float16* pS11 = packW(16, 131, 128);
    _Float16* pS12 = packW(18, 128, 128);
    _Float16* pS13 = packW(20, 128, 128);
    _Float16* pS21 = packW(22, 131, 128);
    _Float16* pS22 = packW(24, 128, 128);
    _Float16* pS23 = packW(26, 128, 256);
    _Float16* pS31 = packW(28, 259, 256);
    _Float16* pS32 = packW(30, 256, 256);
    _Float16* pS33 = packW(32, 256, 512);
    _Float16* pC1  = packW( 2, 512, 256);
    _Float16* pC2  = packW( 4, 256, 256);
    _Float16* pC3  = packW( 6, 256,   1);
    _Float16* pR1  = packW(10, 512, 256);
    _Float16* pR2  = packW(12, 256, 256);
    _Float16* pR3  = packW(14, 256,  70);

    const float* xyz      = (const float*)d_in[0];
    const float* features = (const float*)d_in[1];

    prep_kernel<<<B, 256, 0, stream>>>(xyz, feat4);

    size_t ldsUM = (size_t)(128 * 264 + 128 * 136) * 2;
    up_merge_kernel<<<B * N / 128, 256, ldsUM, stream>>>(
        feat4, features, pUp1, bias(35), pUp2, bias(37), pMg, bias(9), merged);

    fps_kernel<<<B, 32, 0, stream>>>(xyz, 512, 128, nxyz1);
    ballq_kernel<<<(B * 128 + 255) / 256, 256, 0, stream>>>(
        xyz, 512, nxyz1, 128, 0.2f * 0.2f, gidx1);

    size_t ldsSA = (size_t)(128 * 168 + 128 * 264) * 2 + 128 * 4 + 8 * 4;
    sa_mlp_kernel<8><<<B * 128 / 2, 256, ldsSA, stream>>>(
        xyz, 512, merged, nxyz1, gidx1, 128,
        pS11, bias(17), pS12, bias(19), pS13, bias(21), feat1);

    fps_kernel<<<B, 32, 0, stream>>>(nxyz1, 128, 32, nxyz2);
    ballq_kernel<<<(B * 32 + 255) / 256, 256, 0, stream>>>(
        nxyz1, 128, nxyz2, 32, 0.4f * 0.4f, gidx2);

    sa_mlp_kernel<16><<<B * 32 / 2, 256, ldsSA, stream>>>(
        nxyz1, 128, feat1, nxyz2, gidx2, 32,
        pS21, bias(23), pS22, bias(25), pS23, bias(27), feat2);

    size_t ldsS3 = (size_t)(128 * 296 + 128 * 520) * 2;
    sa3_kernel<<<B * 32 / 128, 256, ldsS3, stream>>>(
        nxyz2, feat2, pS31, bias(29), pS32, bias(31), pS33, bias(33), gbuf);

    size_t ldsH = (size_t)(128 * 520 + 2 * 128 * 264) * 2;
    heads_kernel<<<1, 256, ldsH, stream>>>(
        gbuf,
        pC1, bias(3),  pC2, bias(5),  pC3, bias(7),
        pR1, bias(11), pR2, bias(13), pR3, bias(15),
        (float*)d_out);
}
